// SpatialHierarchicalWorldModel_57724360458340
// MI455X (gfx1250) — matrix operand order repaired and checked by reading, not compiled, not run on hardware
//
#include <hip/hip_runtime.h>
#include <hip/hip_bf16.h>
#include <math.h>
#include <stdint.h>

// ---------------- model constants ----------------
#define B_    2
#define T_    32
#define P_    37
#define S_    1184          // T*P
#define D_    768
#define H_    12
#define DH_   64
#define DFF_  3072
#define NL_   6
#define M_    (B_*S_)       // 2368 rows for all dense GEMMs

typedef __bf16 bhalf;
typedef __attribute__((ext_vector_type(16))) __bf16 v16bf;
typedef __attribute__((ext_vector_type(8)))  __bf16 v8bf;
typedef __attribute__((ext_vector_type(8)))  float  v8f;

union FragAB { v16bf v; v8bf h[2]; };

// Async global->LDS 16B copy (per lane), tracked by ASYNCcnt.
// ldsoff = wave-relative LDS byte offset (low 32 bits of a generic __shared__
// pointer), gaddr = 64-bit global address. Both must be 16B aligned.
__device__ __forceinline__ void async_copy16(uint32_t ldsoff, uint64_t gaddr) {
  asm volatile("global_load_async_to_lds_b128 %0, %1, off"
               :: "v"(ldsoff), "v"(gaddr) : "memory");
}
__device__ __forceinline__ uint32_t lds_off(const void* p) {
  return (uint32_t)(uintptr_t)p;
}

// =======================================================================
// f32 -> bf16 conversion (weights, once per call)
// =======================================================================
__global__ __launch_bounds__(256) void f32_to_bf16_kernel(
    const float* __restrict__ src, bhalf* __restrict__ dst, int n) {
  int i = blockIdx.x * 256 + threadIdx.x;
  if (i < n) dst[i] = (bhalf)src[i];
}

// =======================================================================
// Embedding: gather l0/l1/l2/action + level + patch + pos -> x (B,S,D) f32
// =======================================================================
__global__ __launch_bounds__(256) void embed_kernel(
    const int* __restrict__ t0, const int* __restrict__ t1,
    const int* __restrict__ t2, const int* __restrict__ act,
    const float* __restrict__ l0e, const float* __restrict__ l1e,
    const float* __restrict__ l2e, const float* __restrict__ ae,
    const float* __restrict__ le,  const float* __restrict__ pe,
    const float* __restrict__ pos, float* __restrict__ x) {
  int n = blockIdx.x * 256 + threadIdx.x;
  if (n >= B_ * S_ * D_) return;
  int d  = n % D_;
  int bs = n / D_;
  int s  = bs % S_;
  int b  = bs / S_;
  int t  = s / P_;
  int p  = s % P_;
  float e;
  if (p < 4) {
    e = l0e[t0[(b*T_ + t)*4 + p] * D_ + d] + le[0*D_ + d] + pe[p*D_ + d];
  } else if (p < 20) {
    e = l1e[t1[(b*T_ + t)*16 + (p-4)] * D_ + d] + le[1*D_ + d] + pe[(p-4)*D_ + d];
  } else if (p < 36) {
    e = l2e[t2[(b*T_ + t)*16 + (p-20)] * D_ + d] + le[2*D_ + d] + pe[(p-20)*D_ + d];
  } else {
    e = ae[act[b*T_ + t] * D_ + d] + le[3*D_ + d];
  }
  x[n] = e + pos[s*D_ + d];
}

// =======================================================================
// LayerNorm over D=768; one row per 256-thread block (3 elems/thread)
// =======================================================================
__global__ __launch_bounds__(256) void layernorm_kernel(
    const float* __restrict__ x, const float* __restrict__ gamma,
    const float* __restrict__ beta, bhalf* __restrict__ outB,
    float* __restrict__ outF) {
  __shared__ float red[256];
  const int row = blockIdx.x;
  const float* xr = x + (size_t)row * D_;
  float vals[3];
  float s = 0.f;
  #pragma unroll
  for (int i = 0; i < 3; i++) { vals[i] = xr[threadIdx.x + i*256]; s += vals[i]; }
  red[threadIdx.x] = s; __syncthreads();
  for (int off = 128; off > 0; off >>= 1) {
    if (threadIdx.x < off) red[threadIdx.x] += red[threadIdx.x + off];
    __syncthreads();
  }
  const float mean = red[0] * (1.0f / D_);
  __syncthreads();
  float s2 = 0.f;
  #pragma unroll
  for (int i = 0; i < 3; i++) { float dv = vals[i] - mean; s2 += dv * dv; }
  red[threadIdx.x] = s2; __syncthreads();
  for (int off = 128; off > 0; off >>= 1) {
    if (threadIdx.x < off) red[threadIdx.x] += red[threadIdx.x + off];
    __syncthreads();
  }
  const float rstd = rsqrtf(red[0] * (1.0f / D_) + 1e-5f);
  #pragma unroll
  for (int i = 0; i < 3; i++) {
    int d = threadIdx.x + i*256;
    float y = (vals[i] - mean) * rstd * gamma[d] + beta[d];
    if (outB) outB[(size_t)row * D_ + d] = (bhalf)y;
    if (outF) outF[(size_t)row * D_ + d] = y;
  }
}

// =======================================================================
// Dense GEMM: out[m,n] = sum_k A[m,k] * W[n,k] + bias[n]  (+gelu) (+resid)
// A: MxK bf16 row-major; W: NxK bf16 row-major ((out,in) layout).
// Block tile 64(M) x 128(N), K-step 32. Double-buffered LDS filled by
// GLOBAL_LOAD_ASYNC_TO_LDS_B128 (ASYNCcnt): tile i+1 DMA overlaps WMMAs on
// tile i; s_wait_asynccnt 3 + barrier hands off (async loads complete
// in order, so <=3 outstanding => tile i resident for this wave).
// =======================================================================
__global__ __launch_bounds__(256) void gemm_bf16_kernel(
    const bhalf* __restrict__ A, const bhalf* __restrict__ W,
    const float* __restrict__ bias, const float* __restrict__ resid,
    float* __restrict__ outF, bhalf* __restrict__ outB,
    int M, int N, int K, int gelu) {
  __shared__ __attribute__((aligned(16))) bhalf As[2][64][40];   // 80B rows
  __shared__ __attribute__((aligned(16))) bhalf Bs[2][128][40];

  const int tid  = threadIdx.x;
  const int wave = tid >> 5;
  const int lane = tid & 31;
  const int g    = lane >> 4;
  const int ln   = lane & 15;
  const int mtile = blockIdx.x * 64;
  const int ntile = blockIdx.y * 128;
  const int sm = (wave & 3) * 16;      // wave's 16-row strip in block tile
  const int nb = (wave >> 2) * 64;     // wave's 64-col half

  v8f acc[4];
  #pragma unroll
  for (int i = 0; i < 4; i++)
    #pragma unroll
    for (int r = 0; r < 8; r++) acc[i][r] = 0.0f;

  const int ar = tid >> 2, ac = (tid & 3) * 8;   // A: one 16B chunk / thread
  const int br = tid >> 1, bc = (tid & 1) * 16;  // B: two 16B chunks / thread
  const bhalf* agsrc = &A[(size_t)(mtile + ar) * K + ac];
  const bhalf* bgsrc = &W[(size_t)(ntile + br) * K + bc];

  // per-tile async stage: 3 async instructions per wave
  auto issue_tile = [&](int k0, int buf) {
    async_copy16(lds_off(&As[buf][ar][ac]), (uint64_t)(uintptr_t)(agsrc + k0));
    uint32_t bdst = lds_off(&Bs[buf][br][bc]);
    uint64_t bsrc = (uint64_t)(uintptr_t)(bgsrc + k0);
    async_copy16(bdst,      bsrc);
    async_copy16(bdst + 16, bsrc + 16);
  };

  const int nk = K >> 5;
  issue_tile(0, 0);
  for (int i = 0; i < nk; i++) {
    const int buf = i & 1;
    if (i + 1 < nk) {
      issue_tile((i + 1) << 5, buf ^ 1);     // prefetch next tile (other buf)
      asm volatile("s_wait_asynccnt 0x3" ::: "memory");  // tile i resident
    } else {
      asm volatile("s_wait_asynccnt 0x0" ::: "memory");
    }
    __syncthreads();                         // all waves' DMA visible

    FragAB a;   // A frag: row = L%16; k = [16g, 16g+16)
    a.h[0] = *(const v8bf*)&As[buf][sm + ln][16 * g];
    a.h[1] = *(const v8bf*)&As[buf][sm + ln][16 * g + 8];
    #pragma unroll
    for (int nt = 0; nt < 4; nt++) {
      FragAB b; // B frag: col = L%16; k = [16g, 16g+16)
      const bhalf* bp = &Bs[buf][nb + nt * 16 + ln][16 * g];
      b.h[0] = *(const v8bf*)&bp[0];
      b.h[1] = *(const v8bf*)&bp[8];
      acc[nt] = __builtin_amdgcn_wmma_f32_16x16x32_bf16(
          false, a.v, false, b.v, (short)0, acc[nt], false, false);
    }
    __syncthreads();   // reads done before this buffer is DMA'd again (i+2)
  }

  // epilogue: bias (+gelu) (+residual) -> f32 and/or bf16
  #pragma unroll
  for (int nt = 0; nt < 4; nt++) {
    const int gn = ntile + nb + nt * 16 + ln;
    const float bv = bias ? bias[gn] : 0.0f;
    #pragma unroll
    for (int r = 0; r < 8; r++) {
      const int gm = mtile + sm + r + 8 * g;   // C/D: row = r + 8g
      float v = acc[nt][r] + bv;
      if (gelu) v = 0.5f * v * (1.0f + erff(v * 0.70710678118654752f));
      if (resid) v += resid[(size_t)gm * N + gn];
      if (outF) outF[(size_t)gm * N + gn] = v;
      if (outB) outB[(size_t)gm * N + gn] = (bhalf)v;
    }
  }
}

// =======================================================================
// Flash attention: one wave per (b, h, 16-query tile). Per 32-key tile:
// issue async DMA of the 32x64 V tile into per-wave LDS, run QK^T WMMAs +
// analytic hierarchical mask + online softmax (shfl over the 16-lane wave32
// halves) while the DMA flies, then s_wait_asynccnt 0 and run PV WMMAs with
// B-fragments served from LDS. P relayout (C-layout -> A-layout) also via LDS.
// qkv: (B,S,3D) bf16 with q|k|v at col offsets 0|D|2D, head h at h*64.
// =======================================================================
__global__ __launch_bounds__(256) void attention_kernel(
    const bhalf* __restrict__ qkv, bhalf* __restrict__ out) {
  __shared__ __attribute__((aligned(16))) bhalf plds[8][16][40];
  __shared__ __attribute__((aligned(16))) bhalf Vs[8][32][72];   // 144B rows
  const int wave = threadIdx.x >> 5;
  const int lane = threadIdx.x & 31;
  const int g  = lane >> 4;
  const int ln = lane & 15;
  const int wglobal = blockIdx.x * 8 + wave;     // 222 blocks * 8 = 1776 waves
  const int qt = wglobal % (S_ / 16);            // 74 query tiles
  const int h  = (wglobal / (S_ / 16)) % H_;
  const int b  = wglobal / ((S_ / 16) * H_);
  const int q0 = qt * 16;
  const size_t stride = 3 * D_;
  const bhalf* qbase = qkv + (size_t)(b * S_) * stride + h * DH_;
  const bhalf* kbase = qbase + D_;
  const bhalf* vbase = qbase + 2 * D_;

  // Q A-fragments for the two 32-wide k-steps over DH=64
  FragAB qa[2];
  {
    const bhalf* qrow = qbase + (size_t)(q0 + ln) * stride;
    #pragma unroll
    for (int ks = 0; ks < 2; ks++) {
      qa[ks].h[0] = *(const v8bf*)&qrow[ks * 32 + 16 * g];
      qa[ks].h[1] = *(const v8bf*)&qrow[ks * 32 + 16 * g + 8];
    }
  }

  float rowmax[8], rowsum[8];
  v8f o[4];
  #pragma unroll
  for (int r = 0; r < 8; r++) { rowmax[r] = -3.0e38f; rowsum[r] = 0.0f; }
  #pragma unroll
  for (int i = 0; i < 4; i++)
    #pragma unroll
    for (int r = 0; r < 8; r++) o[i][r] = 0.0f;

  int  qi[8], qts[8]; bool qres[8];
  #pragma unroll
  for (int r = 0; r < 8; r++) {
    qi[r]  = q0 + r + 8 * g;
    qres[r] = ((qi[r] % P_) >= 4);
    qts[r] = qi[r] / P_;
  }

  // V tile DMA mapping: 32 rows x 128B = 256 16B-chunks, 8 per lane
  const int vchunk_base = lane;   // chunk c = ii*32 + lane
  const int ktmax = (q0 + 15) >> 5;
  for (int kt = 0; kt <= ktmax; kt++) {
    // ---- kick off V tile (32 keys x 64 dh) DMA into this wave's LDS ----
    #pragma unroll
    for (int ii = 0; ii < 8; ii++) {
      const int c   = ii * 32 + vchunk_base;
      const int row = c >> 3;            // key within tile
      const int col = (c & 7) * 8;       // dh element offset (16B chunks)
      async_copy16(lds_off(&Vs[wave][row][col]),
                   (uint64_t)(uintptr_t)&vbase[(size_t)(kt * 32 + row) * stride + col]);
    }

    // ---- scores: two 16-key subtiles (overlaps the V DMA) ----
    v8f sc[2];
    #pragma unroll
    for (int nt = 0; nt < 2; nt++) {
      const int key = kt * 32 + nt * 16 + ln;
      const bhalf* krow = kbase + (size_t)key * stride;
      v8f s;
      #pragma unroll
      for (int r = 0; r < 8; r++) s[r] = 0.0f;
      #pragma unroll
      for (int ks = 0; ks < 2; ks++) {
        FragAB kb;  // B frag from K row `key`: k = dh = ks*32 + [16g,16g+16)
        kb.h[0] = *(const v8bf*)&krow[ks * 32 + 16 * g];
        kb.h[1] = *(const v8bf*)&krow[ks * 32 + 16 * g + 8];
        s = __builtin_amdgcn_wmma_f32_16x16x32_bf16(
            false, qa[ks].v, false, kb.v, (short)0, s, false, false);
      }
      sc[nt] = s;
    }
    // ---- mask + online softmax ----
    #pragma unroll
    for (int r = 0; r < 8; r++) {
      float pv0, pv1, mx;
      {
        int kj = kt * 32 + ln;
        int pk = kj % P_, tk = kj / P_;
        bool blk = (kj > qi[r]) || (qres[r] && pk >= 4 && pk < 36 && qts[r] > tk);
        pv0 = blk ? -1.0e30f : sc[0][r] * 0.125f;
        kj += 16; pk = kj % P_; tk = kj / P_;
        blk = (kj > qi[r]) || (qres[r] && pk >= 4 && pk < 36 && qts[r] > tk);
        pv1 = blk ? -1.0e30f : sc[1][r] * 0.125f;
        mx = fmaxf(pv0, pv1);
      }
      for (int m = 1; m < 16; m <<= 1) mx = fmaxf(mx, __shfl_xor(mx, m, 32));
      const float nmax  = fmaxf(rowmax[r], mx);
      const float alpha = expf(rowmax[r] - nmax);
      const float p0 = expf(pv0 - nmax);
      const float p1 = expf(pv1 - nmax);
      float part = p0 + p1;
      for (int m = 1; m < 16; m <<= 1) part += __shfl_xor(part, m, 32);
      rowsum[r] = rowsum[r] * alpha + part;
      rowmax[r] = nmax;
      #pragma unroll
      for (int nt = 0; nt < 4; nt++) o[nt][r] *= alpha;
      plds[wave][r + 8 * g][ln]      = (bhalf)p0;   // C-layout -> LDS
      plds[wave][r + 8 * g][16 + ln] = (bhalf)p1;
    }
    // V DMA done (wave-private tile; ASYNCcnt is per-wave, no barrier needed)
    asm volatile("s_wait_asynccnt 0x0" ::: "memory");

    // ---- PV: P (16x32) x V (32x64) ----
    FragAB pa;
    pa.h[0] = *(const v8bf*)&plds[wave][ln][16 * g];
    pa.h[1] = *(const v8bf*)&plds[wave][ln][16 * g + 8];
    #pragma unroll
    for (int nt = 0; nt < 4; nt++) {
      FragAB vb;  // B frag: col dh = nt*16 + ln; k (key in tile) = j + 16g
      const int dh = nt * 16 + ln;
      #pragma unroll
      for (int j = 0; j < 16; j++) {
        vb.v[j] = Vs[wave][j + 16 * g][dh];
      }
      o[nt] = __builtin_amdgcn_wmma_f32_16x16x32_bf16(
          false, pa.v, false, vb.v, (short)0, o[nt], false, false);
    }
  }

  // ---- normalize and store attn (B,S,D) bf16 ----
  #pragma unroll
  for (int nt = 0; nt < 4; nt++) {
    #pragma unroll
    for (int r = 0; r < 8; r++) {
      out[(size_t)(b * S_ + qi[r]) * D_ + h * DH_ + nt * 16 + ln] =
          (bhalf)(o[nt][r] / rowsum[r]);
    }
  }
}

// =======================================================================
// host-side orchestration
// =======================================================================
extern "C" void kernel_launch(void* const* d_in, const int* in_sizes, int n_in,
                              void* d_out, int out_size, void* d_ws, size_t ws_size,
                              hipStream_t stream) {
  (void)in_sizes; (void)n_in; (void)out_size; (void)ws_size;
  const int*   t0    = (const int*)d_in[0];
  const int*   t1    = (const int*)d_in[1];
  const int*   t2    = (const int*)d_in[2];
  const int*   act   = (const int*)d_in[3];
  const float* l0e   = (const float*)d_in[4];
  const float* l1e   = (const float*)d_in[5];
  const float* l2e   = (const float*)d_in[6];
  const float* ae    = (const float*)d_in[7];
  const float* le    = (const float*)d_in[8];
  const float* pe    = (const float*)d_in[9];
  const float* pos   = (const float*)d_in[10];
  const float* ln1_g = (const float*)d_in[11];
  const float* ln1_b = (const float*)d_in[12];
  const float* in_w  = (const float*)d_in[13];
  const float* in_b  = (const float*)d_in[14];
  const float* out_w = (const float*)d_in[15];
  const float* out_b = (const float*)d_in[16];
  const float* ln2_g = (const float*)d_in[17];
  const float* ln2_b = (const float*)d_in[18];
  const float* w1    = (const float*)d_in[19];
  const float* b1    = (const float*)d_in[20];
  const float* w2    = (const float*)d_in[21];
  const float* b2    = (const float*)d_in[22];
  const float* lnf_g = (const float*)d_in[23];
  const float* lnf_b = (const float*)d_in[24];

  uint8_t* ws = (uint8_t*)d_ws;
  size_t off = 0;
  auto wsalloc = [&](size_t bytes) -> void* {
    void* p = ws + off;
    off += (bytes + 255) & ~(size_t)255;
    return p;
  };
  float* X    = (float*)wsalloc((size_t)M_ * D_ * 4);
  bhalf* XN   = (bhalf*)wsalloc((size_t)M_ * D_ * 2);
  bhalf* QKV  = (bhalf*)wsalloc((size_t)M_ * 3 * D_ * 2);
  bhalf* ATT  = (bhalf*)wsalloc((size_t)M_ * D_ * 2);
  bhalf* Hb   = (bhalf*)wsalloc((size_t)M_ * DFF_ * 2);
  bhalf* WIN  = (bhalf*)wsalloc((size_t)NL_ * 3 * D_ * D_ * 2);
  bhalf* WOUT = (bhalf*)wsalloc((size_t)NL_ * D_ * D_ * 2);
  bhalf* W1b  = (bhalf*)wsalloc((size_t)NL_ * DFF_ * D_ * 2);
  bhalf* W2b  = (bhalf*)wsalloc((size_t)NL_ * D_ * DFF_ * 2);

  auto cvt = [&](const float* src, bhalf* dst, int n) {
    f32_to_bf16_kernel<<<(n + 255) / 256, 256, 0, stream>>>(src, dst, n);
  };
  cvt(in_w,  WIN,  NL_ * 3 * D_ * D_);
  cvt(out_w, WOUT, NL_ * D_ * D_);
  cvt(w1,    W1b,  NL_ * DFF_ * D_);
  cvt(w2,    W2b,  NL_ * D_ * DFF_);

  embed_kernel<<<(B_ * S_ * D_ + 255) / 256, 256, 0, stream>>>(
      t0, t1, t2, act, l0e, l1e, l2e, ae, le, pe, pos, X);

  const int attn_blocks = (B_ * H_ * (S_ / 16)) / 8;  // 1776 waves / 8 = 222
  for (int l = 0; l < NL_; l++) {
    layernorm_kernel<<<M_, 256, 0, stream>>>(X, ln1_g + l * D_, ln1_b + l * D_, XN, nullptr);
    gemm_bf16_kernel<<<dim3(M_ / 64, (3 * D_) / 128), 256, 0, stream>>>(
        XN, WIN + (size_t)l * 3 * D_ * D_, in_b + l * 3 * D_,
        nullptr, nullptr, QKV, M_, 3 * D_, D_, 0);
    attention_kernel<<<attn_blocks, 256, 0, stream>>>(QKV, ATT);
    gemm_bf16_kernel<<<dim3(M_ / 64, D_ / 128), 256, 0, stream>>>(
        ATT, WOUT + (size_t)l * D_ * D_, out_b + l * D_,
        X, X, nullptr, M_, D_, D_, 0);
    layernorm_kernel<<<M_, 256, 0, stream>>>(X, ln2_g + l * D_, ln2_b + l * D_, XN, nullptr);
    gemm_bf16_kernel<<<dim3(M_ / 64, DFF_ / 128), 256, 0, stream>>>(
        XN, W1b + (size_t)l * DFF_ * D_, b1 + l * DFF_,
        nullptr, nullptr, Hb, M_, DFF_, D_, 1);
    gemm_bf16_kernel<<<dim3(M_ / 64, D_ / 128), 256, 0, stream>>>(
        Hb, W2b + (size_t)l * D_ * DFF_, b2 + l * D_,
        X, X, nullptr, M_, D_, DFF_, 0);
  }
  layernorm_kernel<<<M_, 256, 0, stream>>>(X, lnf_g, lnf_b, nullptr, (float*)d_out);
}
